// Fca_SE_GatingModule_70007966925068
// MI455X (gfx1250) — compile-verified, matrix operand order checked by reading, not run on hardware
//
#include <hip/hip_runtime.h>
#include <hip/hip_bf16.h>

typedef __attribute__((ext_vector_type(16))) _Float16 v16h;
typedef __attribute__((ext_vector_type(8)))  _Float16 v8h;
typedef __attribute__((ext_vector_type(8)))  float    v8f;

#define B_    256
#define C_    512
#define HW_   196
#define HID_  128
#define BC_   (B_ * C_)
#define NOUT_ (B_ * C_ * HW_)

// ---------------------------------------------------------------------------
// 1) DCT-weighted pooling: s[b,c] = dot(x[b,c,:,:], dct[c,:,:]) over 196 elems
//    One wave32 per (b,c) plane; 49 float4 MACs + shuffle reduction.
// ---------------------------------------------------------------------------
__global__ void fca_pool(const float* __restrict__ x,
                         const float* __restrict__ dctw,
                         float* __restrict__ s_out,
                         _Float16* __restrict__ s16) {
  const int wid  = blockIdx.x * (blockDim.x >> 5) + (threadIdx.x >> 5);
  const int lane = threadIdx.x & 31;
  const int c    = wid & (C_ - 1);

  const float4* __restrict__ xp = (const float4*)(x + (size_t)wid * HW_);
  const float4* __restrict__ wp = (const float4*)(dctw + (size_t)c * HW_);

  float acc = 0.0f;
  #pragma unroll
  for (int i = lane; i < HW_ / 4; i += 32) {   // 49 float4 per plane
    float4 a = xp[i];
    float4 b = wp[i];
    acc += a.x * b.x + a.y * b.y + a.z * b.z + a.w * b.w;
  }
  #pragma unroll
  for (int off = 16; off > 0; off >>= 1)
    acc += __shfl_xor(acc, off, 32);

  if (lane == 0) {
    s_out[wid] = acc;
    s16[wid]   = (_Float16)acc;
  }
}

// ---------------------------------------------------------------------------
// 2) f32 -> f16 weight conversion
// ---------------------------------------------------------------------------
__global__ void fca_cvt(const float* __restrict__ in,
                        _Float16* __restrict__ out, int n) {
  int i = blockIdx.x * blockDim.x + threadIdx.x;
  if (i < n) out[i] = (_Float16)in[i];
}

// ---------------------------------------------------------------------------
// WMMA fragment helpers: 16-bit A 16x32 layout (ISA 7.12.2):
//   lane L: row m = L&15 (both halves), hi = L>>4 selects K chunks
//   a[0..7]  = K = k0 + hi*8 + i
//   a[8..15] = K = k0 + 16 + hi*8 + i
// B 32x16 mirrors with column n = L&15, K-contiguous rows from weight matrix.
// ---------------------------------------------------------------------------
__device__ __forceinline__ v16h load_frag_k(const _Float16* __restrict__ base,
                                            int row, int ld, int k0, int hi) {
  const v8h lo = *(const v8h*)(base + (size_t)row * ld + k0 + hi * 8);
  const v8h hh = *(const v8h*)(base + (size_t)row * ld + k0 + 16 + hi * 8);
  v16h f;
  #pragma unroll
  for (int i = 0; i < 8; ++i) { f[i] = lo[i]; f[8 + i] = hh[i]; }
  return f;
}

// ---------------------------------------------------------------------------
// 3) GEMM1: H = relu(S[256x512] @ W1^T[512x128]) -> f16 [256x128]
//    One wave per 16x16 tile; tiles = 16 x 8 = 128; K loop 512/32 = 16.
// ---------------------------------------------------------------------------
__global__ void fca_gemm1(const _Float16* __restrict__ S,
                          const _Float16* __restrict__ W1,   // [HID_][C_] K-contig
                          _Float16* __restrict__ H) {
  const int wid  = blockIdx.x * (blockDim.x >> 5) + (threadIdx.x >> 5);
  const int lane = threadIdx.x & 31;
  const int tm = wid >> 3;          // 0..15
  const int tn = wid & 7;           // 0..7
  const int mr = tm * 16 + (lane & 15);
  const int nc = tn * 16 + (lane & 15);
  const int hi = lane >> 4;

  v8f acc = {};
  for (int k0 = 0; k0 < C_; k0 += 32) {
    v16h a = load_frag_k(S,  mr, C_, k0, hi);
    v16h b = load_frag_k(W1, nc, C_, k0, hi);
    acc = __builtin_amdgcn_wmma_f32_16x16x32_f16(
        false, a, false, b, (short)0, acc, false, false);
  }
  #pragma unroll
  for (int r = 0; r < 8; ++r) {
    const int row = tm * 16 + r + hi * 8;   // C/D layout: VGPR r -> M = r (+8 hi)
    const float v = acc[r];
    H[(size_t)row * HID_ + nc] = (_Float16)(v > 0.0f ? v : 0.0f);
  }
}

// ---------------------------------------------------------------------------
// 4) GEMM2: raw = H[256x128] @ W2^T[128x512] -> f32 [256x512]
//    tiles = 16 x 32 = 512; K loop 128/32 = 4.
// ---------------------------------------------------------------------------
__global__ void fca_gemm2(const _Float16* __restrict__ H,
                          const _Float16* __restrict__ W2,   // [C_][HID_] K-contig
                          float* __restrict__ raw) {
  const int wid  = blockIdx.x * (blockDim.x >> 5) + (threadIdx.x >> 5);
  const int lane = threadIdx.x & 31;
  const int tm = wid >> 5;          // 0..15
  const int tn = wid & 31;          // 0..31
  const int mr = tm * 16 + (lane & 15);
  const int nc = tn * 16 + (lane & 15);
  const int hi = lane >> 4;

  v8f acc = {};
  #pragma unroll
  for (int k0 = 0; k0 < HID_; k0 += 32) {
    v16h a = load_frag_k(H,  mr, HID_, k0, hi);
    v16h b = load_frag_k(W2, nc, HID_, k0, hi);
    acc = __builtin_amdgcn_wmma_f32_16x16x32_f16(
        false, a, false, b, (short)0, acc, false, false);
  }
  #pragma unroll
  for (int r = 0; r < 8; ++r) {
    const int row = tm * 16 + r + hi * 8;
    raw[(size_t)row * C_ + nc] = acc[r];
  }
}

// ---------------------------------------------------------------------------
// 5) Per-row top-k mask (stable-rank) + tanh. One block per batch row.
//    rank(c) = #{j : raw[j] > raw[c]  or  (raw[j]==raw[c] and j<c)}
//    mask = rank < k[b]   (matches stable argsort(-raw) + scatter of cmp)
// ---------------------------------------------------------------------------
__global__ void fca_topk(const float* __restrict__ raw,
                         const int* __restrict__ kt,
                         float* __restrict__ bounded,
                         float* __restrict__ mask,
                         float* __restrict__ gate) {
  __shared__ float row[C_];
  const int b = blockIdx.x;
  const float* __restrict__ r = raw + (size_t)b * C_;
  for (int c = threadIdx.x; c < C_; c += blockDim.x) row[c] = r[c];
  __syncthreads();

  const int kk = kt[b];
  for (int c = threadIdx.x; c < C_; c += blockDim.x) {
    const float v = row[c];
    int rank = 0;
    #pragma unroll 8
    for (int j = 0; j < C_; ++j) {
      const float u = row[j];
      rank += (u > v) || ((u == v) && (j < c));
    }
    const float mv = (rank < kk) ? 1.0f : 0.0f;
    const size_t o = (size_t)b * C_ + c;
    bounded[o] = tanhf(v);
    mask[o]    = mv;
    gate[o]    = mv;   // forward gate == mask (straight-through)
  }
}

// ---------------------------------------------------------------------------
// 6) out = x * gate[b,c]  (float4 grid-stride; 196 % 4 == 0 so no straddle)
// ---------------------------------------------------------------------------
__global__ void fca_apply(const float* __restrict__ x,
                          const float* __restrict__ gate,
                          float* __restrict__ out) {
  const int n4 = NOUT_ / 4;
  for (int i = blockIdx.x * blockDim.x + threadIdx.x; i < n4;
       i += gridDim.x * blockDim.x) {
    const int bc = (i * 4) / HW_;
    const float g = gate[bc];
    float4 v = ((const float4*)x)[i];
    v.x *= g; v.y *= g; v.z *= g; v.w *= g;
    ((float4*)out)[i] = v;
  }
}

// ---------------------------------------------------------------------------
extern "C" void kernel_launch(void* const* d_in, const int* in_sizes, int n_in,
                              void* d_out, int out_size, void* d_ws, size_t ws_size,
                              hipStream_t stream) {
  const float* x    = (const float*)d_in[0];
  const float* dctw = (const float*)d_in[1];
  const float* w1   = (const float*)d_in[2];
  const float* w2   = (const float*)d_in[3];
  const int*   kt   = (const int*)d_in[4];

  float* out     = (float*)d_out;
  float* bounded = out + NOUT_;
  float* raw     = bounded + BC_;
  float* mask    = raw + BC_;
  float* s       = mask + BC_;

  char* ws = (char*)d_ws;
  size_t off = 0;
  _Float16* s16   = (_Float16*)(ws + off); off += (size_t)BC_ * 2;        // 256 KiB
  _Float16* w1_16 = (_Float16*)(ws + off); off += (size_t)C_ * HID_ * 2;  // 128 KiB
  _Float16* w2_16 = (_Float16*)(ws + off); off += (size_t)C_ * HID_ * 2;  // 128 KiB
  _Float16* h16   = (_Float16*)(ws + off); off += (size_t)B_ * HID_ * 2;  //  64 KiB
  float*    gate  = (float*)(ws + off);                                   // 512 KiB

  // 1) pooling: BC_ waves, 8 waves per block
  fca_pool<<<BC_ / 8, 256, 0, stream>>>(x, dctw, s, s16);
  // 2) weight conversion
  fca_cvt<<<(C_ * HID_ + 255) / 256, 256, 0, stream>>>(w1, w1_16, C_ * HID_);
  fca_cvt<<<(C_ * HID_ + 255) / 256, 256, 0, stream>>>(w2, w2_16, C_ * HID_);
  // 3) GEMM1: 128 tiles, 8 waves/block -> 16 blocks (exact; EXEC all-1s)
  fca_gemm1<<<16, 256, 0, stream>>>(s16, w1_16, h16);
  // 4) GEMM2: 512 tiles -> 64 blocks
  fca_gemm2<<<64, 256, 0, stream>>>(h16, w2_16, raw);
  // 5) top-k mask + tanh: one block per row
  fca_topk<<<B_, 256, 0, stream>>>(raw, kt, bounded, mask, gate);
  // 6) gated output
  fca_apply<<<(NOUT_ / 4 + 255) / 256, 256, 0, stream>>>(x, gate, out);
}
